// VAEDecoder_65300682768931
// MI455X (gfx1250) — matrix operand order repaired and checked by reading, not compiled, not run on hardware
//
#include <hip/hip_runtime.h>
#include <hip/hip_bf16.h>

#define BHW 128
#define HM  13
#define WM  13
#define PP  169
#define DD  256
#define TCC 32
#define RHH 512
#define BPN (BHW*PP)   /* 21632 */

typedef __attribute__((ext_vector_type(16))) __bf16 v16bf;
typedef __attribute__((ext_vector_type(2)))  __bf16 v2bf;
typedef __attribute__((ext_vector_type(8)))  float  v8f;

#if defined(__has_builtin)
#if __has_builtin(__builtin_amdgcn_tensor_load_to_lds)
#define HAVE_TDM 1
#endif
#if __has_builtin(__builtin_amdgcn_cvt_pk_bf16_f32)
#define HAVE_CVTPK 1
#endif
#endif

// pack two f32 into two bf16 (one dword). CVTPK: exact RNE in 1 op.
// Fallback: round-half-up (+0x8000) then v_perm_b32 grabs both high halves
// (3 VALU per pair; differs from RNE only on ties, << bf16 quantization).
__device__ __forceinline__ unsigned int pk2bf(float lo, float hi){
#ifdef HAVE_CVTPK
  union { v2bf v; unsigned int u; } c;
  c.v = __builtin_amdgcn_cvt_pk_bf16_f32(lo, hi);
  return c.u;
#else
  unsigned int ul = __float_as_uint(lo) + 0x8000u;
  unsigned int uh = __float_as_uint(hi) + 0x8000u;
  return __builtin_amdgcn_perm(uh, ul, 0x07060302u);  // [lo.b2 lo.b3 hi.b2 hi.b3]
#endif
}
__device__ __forceinline__ unsigned short f2bf(float f){
  return (unsigned short)((__float_as_uint(f) + 0x8000u) >> 16);
}
__device__ __forceinline__ float gelu_f(float x){
  return 0.5f * x * (1.0f + erff(x * 0.70710678118654752f));
}

#ifdef HAVE_TDM
typedef __attribute__((ext_vector_type(4))) unsigned int u32x4;
typedef __attribute__((ext_vector_type(8))) int i32x8;
typedef __attribute__((ext_vector_type(4))) int i32x4;

// TDM: load a 64-row x 32-col bf16 tile (row stride K elems) into LDS with
// 2-dword padding after each 16-dword row -> landed row stride = 36 ushorts.
__device__ __forceinline__ void tdm_load_b_tile(const unsigned short* gsrc,
                                                unsigned lds_off,
                                                int rem_k, int rem_n, int K)
{
  unsigned long long ga = (unsigned long long)(size_t)gsrc;
  u32x4 g0;
  g0[0] = 1u;                                   // count=1 (user descriptor)
  g0[1] = lds_off;                              // lds_addr
  g0[2] = (unsigned int)ga;                     // global_addr[31:0]
  g0[3] = (unsigned int)(ga >> 32) | (2u << 30);// global_addr[56:32] | type=2
  i32x8 g1;
  g1[0] = (int)0x02D10000u;  // data_size=2B | pad_en | pad_interval=16dw | pad_amount=2dw
  g1[1] = (int)(((unsigned)rem_k & 0xFFFFu) << 16);                       // td0[15:0]
  g1[2] = (int)((((unsigned)rem_k >> 16) & 0xFFFFu) |
                (((unsigned)rem_n & 0xFFFFu) << 16));                     // td0 hi | td1 lo
  g1[3] = (int)((((unsigned)rem_n >> 16) & 0xFFFFu) | (32u << 16));       // td1 hi | tile_dim0=32
  g1[4] = 64;                                   // tile_dim1=64
  g1[5] = (int)(unsigned)K;                     // tensor_dim0_stride[31:0]
  g1[6] = 0;
  g1[7] = 0;
  i32x4 z4 = {0, 0, 0, 0};
  i32x8 z8 = {0, 0, 0, 0, 0, 0, 0, 0};
  __builtin_amdgcn_tensor_load_to_lds(g0, g1, z4, z4, z8, 0);
}
#endif

// ---------------------------------------------------------------------------
// WMMA GEMM:  C[m, coff+n] = act( A_f32[m,:K] . Bbf16[n,:K] + bias[n] )
// 64x64 tile, BK=32, 256 threads = 8 waves. B tile fetched by the TDM.
// ---------------------------------------------------------------------------
__launch_bounds__(256)
__global__ void gemm_wmma(const float* __restrict__ A, const unsigned short* __restrict__ Bw,
                          const float* __restrict__ bias, float* __restrict__ C,
                          int M, int N, int K, int lda, int ldc, int coff, int act)
{
  __shared__ __align__(16) unsigned short As[64][40];
  __shared__ __align__(16) unsigned short Bs[64][36];
  const int tid  = threadIdx.x;
  const int srow = tid >> 2;
  const int scol = (tid & 3) * 8;
  const int m0 = blockIdx.y * 64;
  const int n0 = blockIdx.x * 64;
  const int wv = tid >> 5, lane = tid & 31;
  const int wr = wv >> 1, wc = wv & 1;

  v8f acc0 = {}; v8f acc1 = {};

  const int am = m0 + srow;
  const int amc = (am < M) ? am : (M - 1);           // clamp: OOB rows never stored
  const float* arow = A + (size_t)amc * lda;
#ifndef HAVE_TDM
  const int bn  = n0 + srow;
  const int bnc = (bn < N) ? bn : (N - 1);
  const unsigned short* brow = Bw + (size_t)bnc * K;
#endif

  for (int k0 = 0; k0 < K; k0 += 32) {
    if (k0 + 32 < K)
      __builtin_prefetch(arow + k0 + 32 + scol, 0, 1);

    // ---- stage A: 8 f32 -> 4 packed bf16 dwords -> one b128 DS store ----
    float4 f0 = *(const float4*)(arow + k0 + scol);
    float4 f1 = *(const float4*)(arow + k0 + scol + 4);
    uint4 pk;
    pk.x = pk2bf(f0.x, f0.y);  pk.y = pk2bf(f0.z, f0.w);
    pk.z = pk2bf(f1.x, f1.y);  pk.w = pk2bf(f1.z, f1.w);
    *(uint4*)&As[srow][scol] = pk;

    // ---- stage B: async tensor DMA (wave 0) or vectorized fallback ------
#ifdef HAVE_TDM
    if (wv == 0) {
      tdm_load_b_tile(Bw + (size_t)n0 * K + k0, (unsigned)(size_t)&Bs[0][0],
                      K - k0, N - n0, K);
      __builtin_amdgcn_s_wait_tensorcnt(0);
    }
#else
    {
      const uint4 bq = *(const uint4*)(brow + k0 + scol);
      *(uint2*)&Bs[srow][scol]     = make_uint2(bq.x, bq.y);
      *(uint2*)&Bs[srow][scol + 4] = make_uint2(bq.z, bq.w);
    }
#endif
    __syncthreads();

    union { unsigned int u[8]; v16bf v; } fa, fb0, fb1;
    const int ml = wr * 16 + (lane & 15);
    const int kb = (lane & 16) ? 8 : 0;
#pragma unroll
    for (int i = 0; i < 8; ++i) {
      int kk = ((i & 4) ? 16 : 0) + kb + ((i & 3) << 1);   // ISA A 16x32 bf16 layout
      fa.u[i] = *(const unsigned int*)&As[ml][kk];
    }
    const int nl0 = wc * 32 + (lane & 15);
    const int nl1 = nl0 + 16;
    const int kb2 = (lane & 16) ? 8 : 0;
#pragma unroll
    for (int i = 0; i < 8; ++i) {
      int kk = ((i & 4) ? 16 : 0) + kb2 + ((i & 3) << 1);   // ISA B 32x16 bf16 layout
      fb0.u[i] = *(const unsigned int*)&Bs[nl0][kk];
      fb1.u[i] = *(const unsigned int*)&Bs[nl1][kk];
    }
    acc0 = __builtin_amdgcn_wmma_f32_16x16x32_bf16(false, fa.v, false, fb0.v, (short)0, acc0, false, false);
    acc1 = __builtin_amdgcn_wmma_f32_16x16x32_bf16(false, fa.v, false, fb1.v, (short)0, acc1, false, false);
    __syncthreads();
  }

  const int gmb = m0 + wr * 16 + ((lane & 16) >> 1);  // +8 for lanes 16..31
  const int gn0 = n0 + wc * 32 + (lane & 15);
  const int gn1 = gn0 + 16;
#pragma unroll
  for (int r = 0; r < 8; ++r) {
    int gm = gmb + r;
    if (gm < M) {
      if (gn0 < N) {
        float v = acc0[r] + bias[gn0];
        if (act == 1) v = fmaxf(v, 0.f);
        C[(size_t)gm * ldc + coff + gn0] = v;
      }
      if (gn1 < N) {
        float v = acc1[r] + bias[gn1];
        if (act == 1) v = fmaxf(v, 0.f);
        C[(size_t)gm * ldc + coff + gn1] = v;
      }
    }
  }
}

// ---------------------------------------------------------------------------
// conv2 as im2col WMMA GEMM. Row r = bp*9+pos (3x3 VALID on 5x5 input);
// k = cin*9 + (dy*3+dx), matching c2_w (128,64,3,3) rows. K=576, N=128.
// M = BPN*9 is tile-exact: no row guards needed.
// ---------------------------------------------------------------------------
__launch_bounds__(256)
__global__ void gemm_conv2(const float* __restrict__ conv1act, const unsigned short* __restrict__ Bw,
                           const float* __restrict__ bias, float* __restrict__ C,
                           int N, int K)
{
  __shared__ __align__(16) unsigned short As[64][40];
  __shared__ __align__(16) unsigned short Bs[64][36];
  const int tid  = threadIdx.x;
  const int srow = tid >> 2;
  const int scol = (tid & 3) * 8;
  const int m0 = blockIdx.y * 64;
  const int n0 = blockIdx.x * 64;
  const int wv = tid >> 5, lane = tid & 31;
  const int wr = wv >> 1, wc = wv & 1;

  v8f acc0 = {}; v8f acc1 = {};

  const int am = m0 + srow;
  const int bp = am / 9, pos = am % 9;
  const int oy = pos / 3, ox = pos % 3;
  const float* abase = conv1act + (size_t)bp * 25 * 64;
#ifndef HAVE_TDM
  const int bn  = n0 + srow;
  const int bnc = (bn < N) ? bn : (N - 1);
  const unsigned short* brow = Bw + (size_t)bnc * K;
#endif

  for (int k0 = 0; k0 < K; k0 += 32) {
    unsigned int pk[4];
#pragma unroll
    for (int i = 0; i < 8; i += 2) {
      int k1 = k0 + scol + i, k2 = k1 + 1;
      int c1 = k1 / 9, t1 = k1 % 9;
      int c2 = k2 / 9, t2 = k2 % 9;
      float a0 = abase[((oy + t1 / 3) * 5 + (ox + t1 % 3)) * 64 + c1];
      float a1 = abase[((oy + t2 / 3) * 5 + (ox + t2 % 3)) * 64 + c2];
      pk[i >> 1] = pk2bf(a0, a1);
    }
    *(uint4*)&As[srow][scol] = *(uint4*)pk;

#ifdef HAVE_TDM
    if (wv == 0) {
      tdm_load_b_tile(Bw + (size_t)n0 * K + k0, (unsigned)(size_t)&Bs[0][0],
                      K - k0, N - n0, K);
      __builtin_amdgcn_s_wait_tensorcnt(0);
    }
#else
    {
      const uint4 bq = *(const uint4*)(brow + k0 + scol);
      *(uint2*)&Bs[srow][scol]     = make_uint2(bq.x, bq.y);
      *(uint2*)&Bs[srow][scol + 4] = make_uint2(bq.z, bq.w);
    }
#endif
    __syncthreads();

    union { unsigned int u[8]; v16bf v; } fa, fb0, fb1;
    const int ml = wr * 16 + (lane & 15);
    const int kb = (lane & 16) ? 8 : 0;
#pragma unroll
    for (int i = 0; i < 8; ++i) {
      int kk = ((i & 4) ? 16 : 0) + kb + ((i & 3) << 1);
      fa.u[i] = *(const unsigned int*)&As[ml][kk];
    }
    const int nl0 = wc * 32 + (lane & 15);
    const int nl1 = nl0 + 16;
    const int kb2 = (lane & 16) ? 8 : 0;
#pragma unroll
    for (int i = 0; i < 8; ++i) {
      int kk = ((i & 4) ? 16 : 0) + kb2 + ((i & 3) << 1);
      fb0.u[i] = *(const unsigned int*)&Bs[nl0][kk];
      fb1.u[i] = *(const unsigned int*)&Bs[nl1][kk];
    }
    acc0 = __builtin_amdgcn_wmma_f32_16x16x32_bf16(false, fa.v, false, fb0.v, (short)0, acc0, false, false);
    acc1 = __builtin_amdgcn_wmma_f32_16x16x32_bf16(false, fa.v, false, fb1.v, (short)0, acc1, false, false);
    __syncthreads();
  }

  const int gmb = m0 + wr * 16 + ((lane & 16) >> 1);
  const int gn0 = n0 + wc * 32 + (lane & 15);
  const int gn1 = gn0 + 16;
#pragma unroll
  for (int r = 0; r < 8; ++r) {
    int gm = gmb + r;
    if (gn0 < N) C[(size_t)gm * 128 + gn0] = acc0[r] + bias[gn0];
    if (gn1 < N) C[(size_t)gm * 128 + gn1] = acc1[r] + bias[gn1];
  }
}

// ---------------------------------------------------------------------------
// conv1 via one-hot lookup. NOTE: reference transposes one-hot spatial axes
// but NOT the mask channel: tile read at [jj][ii], mask at [ii][jj].
// ---------------------------------------------------------------------------
__global__ void conv1_kernel(const int* __restrict__ tmap, const float* __restrict__ w,
                             const float* __restrict__ cb, const float* __restrict__ g,
                             const float* __restrict__ bb, float* __restrict__ out)
{
  __shared__ int   tile[25];
  __shared__ float msk[25];
  const int bp = blockIdx.x;
  const int b = bp / PP, p = bp % PP;
  const int y = p / WM,  x = p % WM;
  const int tid = threadIdx.x;
  if (tid < 25) {
    int i = tid / 5, j = tid % 5;
    int r0 = y + i - 4, c0 = x + j - 2;
    bool valid = (r0 >= 0) && (c0 >= 0) && (c0 < WM) && !((i == 4) && (j >= 2));
    tile[tid] = valid ? tmap[b * PP + r0 * WM + c0] : 0;
    msk[tid]  = valid ? 1.f : 0.f;
  }
  __syncthreads();
  const float bnsc = 0.99999500003749969f;   // 1/sqrt(1+1e-5)
  for (int item = tid; item < 1600; item += 256) {
    int pos = item >> 6;      // 0..24 -> (h,w)
    int c   = item & 63;
    int oy = pos / 5, ox = pos % 5;
    float acc = cb[c];
#pragma unroll
    for (int di = 0; di < 3; ++di) {
      int ii = oy + di - 1;
      if (ii < 0 || ii >= 5) continue;
#pragma unroll
      for (int dj = 0; dj < 3; ++dj) {
        int jj = ox + dj - 1;
        if (jj < 0 || jj >= 5) continue;
        int t   = tile[jj * 5 + ii];        // one-hot channels spatially transposed
        float m = msk[ii * 5 + jj];         // mask channel is not
        acc += w[c * 297 + t * 9 + di * 3 + dj] + m * w[c * 297 + 32 * 9 + di * 3 + dj];
      }
    }
    float a = acc * (g[c] * bnsc) + bb[c];
    out[((size_t)bp * 25 + pos) * 64 + c] = gelu_f(a);
  }
}

// BN2 affine + GELU + transpose (bp,pos,c) -> (bp, c*9+pos)
__global__ void conv2_post(const float* __restrict__ raw, const float* __restrict__ g,
                           const float* __restrict__ bb, float* __restrict__ out1152)
{
  const int bp = blockIdx.x, tid = threadIdx.x;
  const float bnsc = 0.99999500003749969f;
  for (int item = tid; item < 1152; item += 256) {
    int c = item / 9, pos = item % 9;
    float v = raw[((size_t)bp * 9 + pos) * 128 + c];
    v = v * (g[c] * bnsc) + bb[c];
    out1152[(size_t)bp * 1152 + item] = gelu_f(v);
  }
}

// pos input assembly: [row_emb[y] | col_emb[x]]  (169 x 512)
__global__ void posin_kernel(const float* __restrict__ row_emb, const float* __restrict__ col_emb,
                             float* __restrict__ posin)
{
  int idx = blockIdx.x * 256 + threadIdx.x;
  if (idx >= PP * 512) return;
  int p = idx >> 9, c = idx & 511;
  posin[idx] = (c < 256) ? row_emb[(p / WM) * 256 + c] : col_emb[(p % WM) * 256 + (c - 256)];
}

// token slots 0..2 (slot 3 written by patch_fc GEMM with coff=768)
__global__ void assemble_tok(const int* __restrict__ tmap, const float* __restrict__ tile_emb,
                             const float* __restrict__ mv, const float* __restrict__ pos,
                             float* __restrict__ tok)
{
  const int bp = blockIdx.x;
  const int b = bp / PP, p = bp % PP;
  const int tid = threadIdx.x;
  const int prev = (p == 0) ? 31 : tmap[b * PP + p - 1];
  size_t base = (size_t)bp * 4 * DD;
  tok[base + tid]          = tile_emb[prev * DD + tid];
  tok[base + DD + tid]     = mv[b * DD + tid];
  tok[base + 2 * DD + tid] = pos[p * DD + tid];
}

// attention: one wave per (bp, head); 4 tokens x 128 dims, softmax over 4
__global__ void attn_kernel(const float* __restrict__ qkv, float* __restrict__ o)
{
  const int w = threadIdx.x >> 5, lane = threadIdx.x & 31;
  const int unit = blockIdx.x * 8 + w;
  const int bp = unit >> 1, head = unit & 1;
  const float* base = qkv + (size_t)bp * 4 * 768 + head * 128;
  float q[4][4], k[4][4], v[4][4];
#pragma unroll
  for (int t = 0; t < 4; ++t)
#pragma unroll
    for (int j = 0; j < 4; ++j) {
      int d = j * 32 + lane;
      q[t][j] = base[t * 768 + d];
      k[t][j] = base[t * 768 + 256 + d];
      v[t][j] = base[t * 768 + 512 + d];
    }
  float sc[4][4];
#pragma unroll
  for (int qi = 0; qi < 4; ++qi)
#pragma unroll
    for (int ki = 0; ki < 4; ++ki) {
      float s = q[qi][0] * k[ki][0] + q[qi][1] * k[ki][1] + q[qi][2] * k[ki][2] + q[qi][3] * k[ki][3];
      s += __shfl_xor(s, 1, 32);  s += __shfl_xor(s, 2, 32);
      s += __shfl_xor(s, 4, 32);  s += __shfl_xor(s, 8, 32);
      s += __shfl_xor(s, 16, 32);
      sc[qi][ki] = s * 0.088388347648318447f;  // 1/sqrt(128)
    }
#pragma unroll
  for (int qi = 0; qi < 4; ++qi) {
    float mx = fmaxf(fmaxf(sc[qi][0], sc[qi][1]), fmaxf(sc[qi][2], sc[qi][3]));
    float e0 = expf(sc[qi][0] - mx), e1 = expf(sc[qi][1] - mx);
    float e2 = expf(sc[qi][2] - mx), e3 = expf(sc[qi][3] - mx);
    float inv = 1.f / (e0 + e1 + e2 + e3);
    float a0 = e0 * inv, a1 = e1 * inv, a2 = e2 * inv, a3 = e3 * inv;
#pragma unroll
    for (int j = 0; j < 4; ++j) {
      float ov = a0 * v[0][j] + a1 * v[1][j] + a2 * v[2][j] + a3 * v[3][j];
      o[(size_t)(bp * 4 + qi) * 256 + head * 128 + j * 32 + lane] = ov;
    }
  }
}

// LayerNorm: Y = gelu?( LN(X + R) * g + b ), one block per row, L <= 512
__global__ void ln_kernel(const float* __restrict__ X, const float* __restrict__ R,
                          float* __restrict__ Y, const float* __restrict__ g,
                          const float* __restrict__ bta, int L, int do_gelu)
{
  __shared__ float red[256];
  const int row = blockIdx.x, tid = threadIdx.x;
  const float* xr = X + (size_t)row * L;
  const float* rr = R ? (R + (size_t)row * L) : nullptr;
  float v0 = 0.f, v1 = 0.f;
  if (tid < L)       v0 = xr[tid]       + (rr ? rr[tid]       : 0.f);
  if (tid + 256 < L) v1 = xr[tid + 256] + (rr ? rr[tid + 256] : 0.f);
  red[tid] = v0 + v1; __syncthreads();
  for (int o = 128; o > 0; o >>= 1) { if (tid < o) red[tid] += red[tid + o]; __syncthreads(); }
  float mean = red[0] / (float)L; __syncthreads();
  float d0 = (tid < L) ? v0 - mean : 0.f;
  float d1 = (tid + 256 < L) ? v1 - mean : 0.f;
  red[tid] = d0 * d0 + d1 * d1; __syncthreads();
  for (int o = 128; o > 0; o >>= 1) { if (tid < o) red[tid] += red[tid + o]; __syncthreads(); }
  float rs = rsqrtf(red[0] / (float)L + 1e-5f);
  if (tid < L) {
    float y = d0 * rs * g[tid] + bta[tid];
    if (do_gelu) y = gelu_f(y);
    Y[(size_t)row * L + tid] = y;
  }
  if (tid + 256 < L) {
    float y = d1 * rs * g[tid + 256] + bta[tid + 256];
    if (do_gelu) y = gelu_f(y);
    Y[(size_t)row * L + tid + 256] = y;
  }
}

// feat = h[:,0] reordered (b,p) -> (p,b) for the GRU scan
__global__ void featT_kernel(const float* __restrict__ h, float* __restrict__ featT)
{
  const int row = blockIdx.x;          // p*B + b
  const int p = row / BHW, b = row % BHW;
  featT[(size_t)row * DD + threadIdx.x] = h[(size_t)((b * PP + p) * 4) * DD + threadIdx.x];
}

__global__ void zero_kernel(float* __restrict__ p, int n)
{
  int i = blockIdx.x * 256 + threadIdx.x;
  if (i < n) p[i] = 0.f;
}

// packed f32 -> bf16 weight conversion (all weight element counts are even)
__global__ void cvt_bf16_kernel(const float* __restrict__ src, unsigned short* __restrict__ dst, int n)
{
  int i = (blockIdx.x * 256 + threadIdx.x) * 2;
  if (i < n) *(unsigned int*)(dst + i) = pk2bf(src[i], src[i + 1]);
}

__global__ void gru_gate(const float* __restrict__ gi, const float* __restrict__ gh,
                         float* __restrict__ hid, float* __restrict__ hs, int t)
{
  const int idx = blockIdx.x * 256 + threadIdx.x;  // 128*512
  const int b = idx >> 9, j = idx & 511;
  const float* gir = gi + (size_t)(t * BHW + b) * 1536;
  const float* ghr = gh + (size_t)b * 1536;
  float r = 1.f / (1.f + expf(-(gir[j]        + ghr[j])));
  float z = 1.f / (1.f + expf(-(gir[512 + j]  + ghr[512 + j])));
  float n = tanhf(gir[1024 + j] + r * ghr[1024 + j]);
  float hnew = (1.f - z) * n + z * hid[idx];
  hid[idx] = hnew;
  hs[(size_t)(t * BHW + b) * RHH + j] = hnew;
}

// logits rows are (t*B+b); out is (B, TC, H, W) with p == t
__global__ void final_remap(const float* __restrict__ logits, float* __restrict__ out)
{
  const int idx = blockIdx.x * 256 + threadIdx.x;   // BPN*32
  const int row = idx >> 5, c = idx & 31;
  const int t = row / BHW, b = row % BHW;
  out[((size_t)b * TCC + c) * PP + t] = logits[(size_t)row * TCC + c];
}

// ---------------------------------------------------------------------------
extern "C" void kernel_launch(void* const* d_in, const int* in_sizes, int n_in,
                              void* d_out, int out_size, void* d_ws, size_t ws_size,
                              hipStream_t stream)
{
  (void)in_sizes; (void)n_in; (void)out_size; (void)ws_size;
  const float* map_vec    = (const float*)d_in[0];
  const int*   target_map = (const int*)  d_in[1];
  const float* mv_w  = (const float*)d_in[2];
  const float* mv_b  = (const float*)d_in[3];
  const float* tile_emb = (const float*)d_in[4];
  const float* row_emb  = (const float*)d_in[5];
  const float* col_emb  = (const float*)d_in[6];
  const float* pos_w = (const float*)d_in[7];
  const float* pos_b = (const float*)d_in[8];
  const float* c1_w = (const float*)d_in[9];
  const float* c1_b = (const float*)d_in[10];
  const float* bn1_g = (const float*)d_in[11];
  const float* bn1_b = (const float*)d_in[12];
  const float* c2_w = (const float*)d_in[13];
  const float* c2_b = (const float*)d_in[14];
  const float* bn2_g = (const float*)d_in[15];
  const float* bn2_b = (const float*)d_in[16];
  const float* patch_fc_w = (const float*)d_in[17];
  const float* patch_fc_b = (const float*)d_in[18];
  const float* qkv_w = (const float*)d_in[19];
  const float* qkv_b = (const float*)d_in[20];
  const float* out_w = (const float*)d_in[21];
  const float* out_b = (const float*)d_in[22];
  const float* ln1_g = (const float*)d_in[23];
  const float* ln1_b = (const float*)d_in[24];
  const float* ff1_w = (const float*)d_in[25];
  const float* ff1_b = (const float*)d_in[26];
  const float* ff2_w = (const float*)d_in[27];
  const float* ff2_b = (const float*)d_in[28];
  const float* ln2_g = (const float*)d_in[29];
  const float* ln2_b = (const float*)d_in[30];
  const float* gru_wih = (const float*)d_in[31];
  const float* gru_whh = (const float*)d_in[32];
  const float* gru_bih = (const float*)d_in[33];
  const float* gru_bhh = (const float*)d_in[34];
  const float* fc1_w = (const float*)d_in[35];
  const float* fc1_b = (const float*)d_in[36];
  const float* fcln_g = (const float*)d_in[37];
  const float* fcln_b = (const float*)d_in[38];
  const float* fc2_w = (const float*)d_in[39];
  const float* fc2_b = (const float*)d_in[40];

  size_t off = 0;
  auto allocb = [&](size_t bytes) -> void* {
    void* p = (char*)d_ws + off;
    off += (bytes + 255) & ~(size_t)255;
    return p;
  };
  auto allocf = [&](size_t n) -> float* { return (float*)allocb(n * 4); };
  auto alloch = [&](size_t n) -> unsigned short* { return (unsigned short*)allocb(n * 2); };

  float* posin    = allocf((size_t)PP * 512);
  float* posout   = allocf((size_t)PP * DD);
  float* mvbuf    = allocf((size_t)BHW * DD);
  float* tok      = allocf((size_t)BPN * 4 * DD);      // also "h" through transformer
  float* conv1act = allocf((size_t)BPN * 25 * 64);
  float* conv2raw = allocf((size_t)BPN * 9 * 128);
  float* pf1152   = allocf((size_t)BPN * 1152);
  float* qkvbuf   = allocf((size_t)BPN * 4 * 768);
  float* obuf     = allocf((size_t)BPN * 4 * DD);
  float* projbuf  = allocf((size_t)BPN * 4 * DD);
  float* fftmp    = allocf((size_t)BPN * 4 * DD);
  float* featT    = allocf((size_t)BPN * DD);
  float* gibuf    = allocf((size_t)BPN * 1536);
  float* hid      = allocf((size_t)BHW * RHH);
  float* ghbuf    = allocf((size_t)BHW * 1536);
  float* hsbuf    = allocf((size_t)BPN * RHH);
  float* ybuf     = allocf((size_t)BPN * RHH);
  float* logitbuf = allocf((size_t)BPN * TCC);

  // bf16 weight copies (fed to WMMA B / TDM)
  unsigned short* wb_pos = alloch(256 * 512);
  unsigned short* wb_mv  = alloch(256 * 32);
  unsigned short* wb_c2  = alloch(128 * 576);
  unsigned short* wb_pfc = alloch(256 * 1152);
  unsigned short* wb_qkv = alloch((size_t)3 * 768 * 256);
  unsigned short* wb_out = alloch((size_t)3 * 256 * 256);
  unsigned short* wb_ff1 = alloch((size_t)3 * 256 * 256);
  unsigned short* wb_ff2 = alloch((size_t)3 * 256 * 256);
  unsigned short* wb_gih = alloch(1536 * 256);
  unsigned short* wb_ghh = alloch(1536 * 512);
  unsigned short* wb_fc1 = alloch(512 * 512);
  unsigned short* wb_fc2 = alloch(32 * 512);

  auto cvt = [&](const float* s, unsigned short* d, int n) {
    cvt_bf16_kernel<<<(n / 2 + 255) / 256, 256, 0, stream>>>(s, d, n);
  };
  cvt(pos_w, wb_pos, 256 * 512);
  cvt(mv_w, wb_mv, 256 * 32);
  cvt(c2_w, wb_c2, 128 * 576);
  cvt(patch_fc_w, wb_pfc, 256 * 1152);
  cvt(qkv_w, wb_qkv, 3 * 768 * 256);
  cvt(out_w, wb_out, 3 * 256 * 256);
  cvt(ff1_w, wb_ff1, 3 * 256 * 256);
  cvt(ff2_w, wb_ff2, 3 * 256 * 256);
  cvt(gru_wih, wb_gih, 1536 * 256);
  cvt(gru_whh, wb_ghh, 1536 * 512);
  cvt(fc1_w, wb_fc1, 512 * 512);
  cvt(fc2_w, wb_fc2, 32 * 512);

  auto gemm = [&](const float* A, const unsigned short* Bw, const float* bias, float* C,
                  int M, int N, int K, int lda, int ldc, int coff, int act) {
    dim3 g((N + 63) / 64, (M + 63) / 64);
    gemm_wmma<<<g, 256, 0, stream>>>(A, Bw, bias, C, M, N, K, lda, ldc, coff, act);
  };

  // ---- embeddings / projections -----------------------------------------
  posin_kernel<<<(PP * 512) / 256, 256, 0, stream>>>(row_emb, col_emb, posin);
  gemm(posin, wb_pos, pos_b, posout, PP, DD, 512, 512, DD, 0, 0);
  gemm(map_vec, wb_mv, mv_b, mvbuf, BHW, DD, 32, 32, DD, 0, 0);
  assemble_tok<<<BPN, 256, 0, stream>>>(target_map, tile_emb, mvbuf, posout, tok);

  // ---- conv stack + patch fc --------------------------------------------
  conv1_kernel<<<BPN, 256, 0, stream>>>(target_map, c1_w, c1_b, bn1_g, bn1_b, conv1act);
  {
    dim3 g(2, (BPN * 9) / 64);
    gemm_conv2<<<g, 256, 0, stream>>>(conv1act, wb_c2, c2_b, conv2raw, 128, 576);
  }
  conv2_post<<<BPN, 256, 0, stream>>>(conv2raw, bn2_g, bn2_b, pf1152);
  gemm(pf1152, wb_pfc, patch_fc_b, tok, BPN, DD, 1152, 1152, 4 * DD, 3 * DD, 0);

  // ---- transformer (3 layers) -------------------------------------------
  const int MT = BPN * 4;
  for (int l = 0; l < 3; ++l) {
    gemm(tok, wb_qkv + (size_t)l * 768 * 256, qkv_b + l * 768, qkvbuf, MT, 768, 256, 256, 768, 0, 0);
    attn_kernel<<<(BPN * 2) / 8, 256, 0, stream>>>(qkvbuf, obuf);
    gemm(obuf, wb_out + (size_t)l * 256 * 256, out_b + l * 256, projbuf, MT, 256, 256, 256, 256, 0, 0);
    ln_kernel<<<MT, 256, 0, stream>>>(projbuf, tok, tok, ln1_g + l * 256, ln1_b + l * 256, 256, 0);
    gemm(tok, wb_ff1 + (size_t)l * 256 * 256, ff1_b + l * 256, fftmp, MT, 256, 256, 256, 256, 0, 1);
    gemm(fftmp, wb_ff2 + (size_t)l * 256 * 256, ff2_b + l * 256, projbuf, MT, 256, 256, 256, 256, 0, 0);
    ln_kernel<<<MT, 256, 0, stream>>>(projbuf, tok, tok, ln2_g + l * 256, ln2_b + l * 256, 256, 0);
  }

  // ---- GRU ---------------------------------------------------------------
  featT_kernel<<<BPN, 256, 0, stream>>>(tok, featT);
  gemm(featT, wb_gih, gru_bih, gibuf, BPN, 1536, 256, 256, 1536, 0, 0);
  zero_kernel<<<(BHW * RHH) / 256, 256, 0, stream>>>(hid, BHW * RHH);
  for (int t = 0; t < PP; ++t) {
    gemm(hid, wb_ghh, gru_bhh, ghbuf, BHW, 1536, 512, 512, 1536, 0, 0);
    gru_gate<<<(BHW * RHH) / 256, 256, 0, stream>>>(gibuf, ghbuf, hid, hsbuf, t);
  }

  // ---- head --------------------------------------------------------------
  gemm(hsbuf, wb_fc1, fc1_b, ybuf, BPN, 512, 512, 512, 512, 0, 0);
  ln_kernel<<<BPN, 256, 0, stream>>>(ybuf, nullptr, ybuf, fcln_g, fcln_b, 512, 1);
  gemm(ybuf, wb_fc2, fc2_b, logitbuf, BPN, TCC, 512, 512, TCC, 0, 0);
  final_remap<<<(BPN * TCC) / 256, 256, 0, stream>>>(logitbuf, (float*)d_out);
}
